// FlashAttentionSimulator_48576080118374
// MI455X (gfx1250) — matrix operand order repaired and checked by reading, not compiled, not run on hardware
//
#include <hip/hip_runtime.h>
#include <hip/hip_bf16.h>

typedef __bf16 bf16_t;
typedef __attribute__((ext_vector_type(16))) __bf16 v16bf;
typedef __attribute__((ext_vector_type(8)))  __bf16 v8bf;
typedef __attribute__((ext_vector_type(8)))  float   v8f;
typedef __attribute__((ext_vector_type(4)))  int     v4i;

#define D_MODEL 1024
#define T_SEQ   2048
#define NHEAD   16
#define HDIM    64

// ---------------- CDNA5 async-copy-to-LDS (ASYNCcnt) with sync fallback ----
#if defined(__has_builtin)
#  if __has_builtin(__builtin_amdgcn_global_load_async_to_lds_b128)
#    define HAS_ASYNC_LDS 1
#  endif
#endif
#ifndef HAS_ASYNC_LDS
#  define HAS_ASYNC_LDS 0
#endif

#if HAS_ASYNC_LDS
#  if __has_builtin(__builtin_amdgcn_s_wait_asynccnt)
#    define ASYNC_WAIT(n) __builtin_amdgcn_s_wait_asynccnt(n)
#  else
#    define ASYNC_WAIT(n) asm volatile("s_wait_asynccnt " #n ::: "memory")
#  endif
#else
#  define ASYNC_WAIT(n) ((void)0)
#endif

static __device__ inline void cp16_to_lds(const bf16_t* g, bf16_t* l) {
#if HAS_ASYNC_LDS
    __builtin_amdgcn_global_load_async_to_lds_b128(
        (__attribute__((address_space(1))) v4i*)(__attribute__((address_space(1))) void*)(g),
        (__attribute__((address_space(3))) v4i*)(__attribute__((address_space(3))) void*)(l),
        0, 0);
#else
    *(v8bf*)l = *(const v8bf*)g;
#endif
}

// ---------------- WMMA fragment helpers -----------------------------------
union FragU { v16bf f; v8bf h[2]; };

// A-fragment (16x32 bf16): lane(half, m=lane&15) holds row m,
// elements j=0..7 -> K = half*8 + j ; j=8..15 -> K = 16 + half*8 + (j-8)
static __device__ inline v16bf load_fragA(const bf16_t* row, int half) {
    FragU u;
    u.h[0] = *(const v8bf*)(row + half * 8);
    u.h[1] = *(const v8bf*)(row + 16 + half * 8);
    return u.f;
}

// B-fragment (32x16 bf16): lane n=lane&15 holds column n,
// elements j=0..7 -> K = half*8 + j ; j=8..15 -> K = 16 + half*8 + (j-8)
static __device__ inline v16bf load_fragB(const bf16_t* col, int half) {
    FragU u;
    u.h[0] = *(const v8bf*)(col + half * 8);
    u.h[1] = *(const v8bf*)(col + 16 + half * 8);
    return u.f;
}

static __device__ inline v8f wmma_bf16(v16bf a, v16bf b, v8f c) {
    return __builtin_amdgcn_wmma_f32_16x16x32_bf16(
        false, a, false, b, (short)0, c, false, false);
}

// ---------------------------------------------------------------- convert
__global__ __launch_bounds__(256) void cvt_f32_bf16(const float* __restrict__ src,
                                                    bf16_t* __restrict__ dst, int n) {
    int i = blockIdx.x * blockDim.x + threadIdx.x;
    if (i < n) dst[i] = (bf16_t)src[i];
}

// ---------------------------------------------------------------- GEMM
// C[8192,1024] = A[8192,1024](bf16) * W^T, W bf16 [1024,1024] row-major.
// mode 0: bf16 [B,H,T,D] | mode 1: bf16 [B,H,D,T] | mode 2: f32 [M,N]+bias
// Block: 256 thr = 8 waves (2M x 4N); block tile 64x256; wave tile 32x64.
// Double-buffered async staging: A tile 64x32, B tile 256x32 per k-step.
__global__ __launch_bounds__(256)
void gemm_bf16_wmma(const bf16_t* __restrict__ A, const bf16_t* __restrict__ W,
                    void* __restrict__ out, const float* __restrict__ bias, int mode) {
    __shared__ __align__(16) bf16_t As[2][64][40];    // padded rows: 80B, conflict-free
    __shared__ __align__(16) bf16_t Bs[2][256][40];

    const int tid  = threadIdx.x;
    const int lane = tid & 31, wave = tid >> 5;
    const int half = lane >> 4, ln = lane & 15;
    const int wm = wave & 1, wn = wave >> 1;
    const int mb = blockIdx.x * 64;
    const int nb = blockIdx.y * 256;

    // copy duties: A = 64 rows x 4 chunks (1/thread); B = 256 rows x 4 chunks (4/thread)
    const int arow = tid >> 2, achk = tid & 3;
    const bf16_t* gA  = A + (size_t)(mb + arow) * D_MODEL + achk * 8;
    const bf16_t* gB  = W + (size_t)(nb + arow) * D_MODEL + achk * 8;   // rows arow+{0,64,128,192}

    v8f acc[2][4] = {};

    auto issue = [&](int buf, int k0) {
        cp16_to_lds(gA + k0, &As[buf][arow][achk * 8]);
#pragma unroll
        for (int r = 0; r < 4; ++r)
            cp16_to_lds(gB + (size_t)(r * 64) * D_MODEL + k0, &Bs[buf][r * 64 + arow][achk * 8]);
    };

    issue(0, 0);
    const int NK = D_MODEL / 32;
    for (int ks = 0; ks < NK; ++ks) {
        const int buf = ks & 1;
        if (ks + 1 < NK) { issue(buf ^ 1, (ks + 1) * 32); ASYNC_WAIT(5); }
        else             { ASYNC_WAIT(0); }
        __syncthreads();

        v16bf fa0 = load_fragA(&As[buf][wm * 32 + ln][0], half);
        v16bf fa1 = load_fragA(&As[buf][wm * 32 + 16 + ln][0], half);
#pragma unroll
        for (int j = 0; j < 4; ++j) {
            v16bf fb = load_fragB(&Bs[buf][wn * 64 + j * 16 + ln][0], half);
            acc[0][j] = wmma_bf16(fa0, fb, acc[0][j]);
            acc[1][j] = wmma_bf16(fa1, fb, acc[1][j]);
        }
        __syncthreads();
    }

#pragma unroll
    for (int i = 0; i < 2; ++i)
#pragma unroll
        for (int j = 0; j < 4; ++j)
#pragma unroll
            for (int g = 0; g < 8; ++g) {
                const int row = mb + wm * 32 + i * 16 + g + half * 8; // token index b*T+t
                const int col = nb + wn * 64 + j * 16 + ln;           // feature h*64+d
                const float v = acc[i][j][g];
                if (mode == 0) {
                    const int b = row >> 11, t = row & 2047, h = col >> 6, d = col & 63;
                    ((bf16_t*)out)[((((size_t)b * NHEAD + h) * T_SEQ) + t) * HDIM + d] = (bf16_t)v;
                } else if (mode == 1) {
                    const int b = row >> 11, t = row & 2047, h = col >> 6, d = col & 63;
                    ((bf16_t*)out)[(((size_t)b * NHEAD + h) * HDIM + d) * T_SEQ + t] = (bf16_t)v;
                } else {
                    ((float*)out)[(size_t)row * D_MODEL + col] = v + bias[col];
                }
            }
}

// ---------------------------------------------------------------- flash attention
// Q,K: [B,H,T,64] bf16 ; Vt: [B,H,64,T] bf16 ; attn out: [B*T,1024] bf16.
// Block = 8 waves sharing one (b,h); each wave owns 16 query rows.
// K tile (32 keys x 64d) and V tile (64d x 32 keys) double-buffered in LDS.
__global__ __launch_bounds__(256)
void flash_attn_wmma(const bf16_t* __restrict__ Q, const bf16_t* __restrict__ Kb,
                     const bf16_t* __restrict__ Vt, bf16_t* __restrict__ attn) {
    __shared__ __align__(16) bf16_t Ks[2][32][72];    // 144B rows: conflict-free
    __shared__ __align__(16) bf16_t Vs[2][64][40];    // 80B rows
    __shared__ __align__(16) bf16_t ldsP[8][16][40];  // per-wave P tile

    const int tid  = threadIdx.x;
    const int lane = tid & 31, wave = tid >> 5;
    const int half = lane >> 4, ln = lane & 15;
    const int bh = blockIdx.x;
    const int b = bh >> 4, h = bh & 15;
    const int q0 = blockIdx.y * 128 + wave * 16;

    const bf16_t* kbh = Kb + (size_t)bh * T_SEQ * HDIM;
    const bf16_t* vbh = Vt + (size_t)bh * HDIM * T_SEQ;

    // copy duties: K = 32 rows x 8 chunks (1/thread); V = 64 rows x 4 chunks (1/thread)
    const int krow = tid >> 3, kchk = tid & 7;
    const int vrow = tid >> 2, vchk = tid & 3;
    auto issue = [&](int buf, int kb0) {
        cp16_to_lds(kbh + (size_t)(kb0 + krow) * HDIM + kchk * 8, &Ks[buf][krow][kchk * 8]);
        cp16_to_lds(vbh + (size_t)vrow * T_SEQ + kb0 + vchk * 8, &Vs[buf][vrow][vchk * 8]);
    };

    const bf16_t* qrow = Q + ((size_t)bh * T_SEQ + q0 + ln) * HDIM;
    v16bf qa0 = load_fragA(qrow, half);
    v16bf qa1 = load_fragA(qrow + 32, half);

    v8f o[4] = {};
    float mrow[8], lrow[8];
#pragma unroll
    for (int g = 0; g < 8; ++g) { mrow[g] = -3.0e38f; lrow[g] = 0.0f; }

    issue(0, 0);
    for (int kb0 = 0; kb0 < T_SEQ; kb0 += 32) {
        const int buf = (kb0 >> 5) & 1;
        if (kb0 + 32 < T_SEQ) { issue(buf ^ 1, kb0 + 32); ASYNC_WAIT(2); }
        else                  { ASYNC_WAIT(0); }
        __syncthreads();

        // ---- S = (Q K^T) / 8 : two 16-key tiles
        v8f s[2] = {};
#pragma unroll
        for (int t = 0; t < 2; ++t) {
            const bf16_t* kcol = &Ks[buf][t * 16 + ln][0];
            s[t] = wmma_bf16(qa0, load_fragB(kcol, half), s[t]);
            s[t] = wmma_bf16(qa1, load_fragB(kcol + 32, half), s[t]);
        }

        // ---- online softmax (row = g + 8*half)
        float scal[8];
#pragma unroll
        for (int g = 0; g < 8; ++g) {
            s[0][g] *= 0.125f;
            s[1][g] *= 0.125f;
            float t = fmaxf(s[0][g], s[1][g]);
            t = fmaxf(t, __shfl_xor(t, 1, 32));
            t = fmaxf(t, __shfl_xor(t, 2, 32));
            t = fmaxf(t, __shfl_xor(t, 4, 32));
            t = fmaxf(t, __shfl_xor(t, 8, 32));
            const float mnew = fmaxf(mrow[g], t);
            scal[g] = __expf(mrow[g] - mnew);
            const float p0 = __expf(s[0][g] - mnew);
            const float p1 = __expf(s[1][g] - mnew);
            s[0][g] = p0;
            s[1][g] = p1;
            float r = p0 + p1;
            r += __shfl_xor(r, 1, 32);
            r += __shfl_xor(r, 2, 32);
            r += __shfl_xor(r, 4, 32);
            r += __shfl_xor(r, 8, 32);
            lrow[g] = lrow[g] * scal[g] + r;
            mrow[g] = mnew;
        }
#pragma unroll
        for (int t = 0; t < 4; ++t)
#pragma unroll
            for (int g = 0; g < 8; ++g) o[t][g] *= scal[g];

        // ---- P (C layout) -> per-wave LDS -> A fragment (same-wave DS is in-order)
#pragma unroll
        for (int t = 0; t < 2; ++t)
#pragma unroll
            for (int g = 0; g < 8; ++g)
                ldsP[wave][g + half * 8][t * 16 + ln] = (bf16_t)s[t][g];
        v16bf pa = load_fragA(&ldsP[wave][ln][0], half);

        // ---- O += P * V
#pragma unroll
        for (int t = 0; t < 4; ++t)
            o[t] = wmma_bf16(pa, load_fragB(&Vs[buf][t * 16 + ln][0], half), o[t]);

        __syncthreads();
    }

#pragma unroll
    for (int t = 0; t < 4; ++t)
#pragma unroll
        for (int g = 0; g < 8; ++g) {
            const int row = q0 + g + half * 8;
            const int col = h * HDIM + t * 16 + ln;
            const float v = o[t][g] / (lrow[g] + 1e-8f);
            attn[((size_t)b * T_SEQ + row) * D_MODEL + col] = (bf16_t)v;
        }
}

// ---------------------------------------------------------------- launch
extern "C" void kernel_launch(void* const* d_in, const int* in_sizes, int n_in,
                              void* d_out, int out_size, void* d_ws, size_t ws_size,
                              hipStream_t stream) {
    const float* x  = (const float*)d_in[0];
    const float* Wq = (const float*)d_in[1];
    const float* Wk = (const float*)d_in[2];
    const float* Wv = (const float*)d_in[3];
    const float* Wo = (const float*)d_in[4];
    const float* bo = (const float*)d_in[5];
    float* out = (float*)d_out;

    const size_t M = 8192;  // B*T
    char* p = (char*)d_ws;
    bf16_t* xh  = (bf16_t*)p; p += M * D_MODEL * 2;
    bf16_t* wqh = (bf16_t*)p; p += (size_t)D_MODEL * D_MODEL * 2;
    bf16_t* wkh = (bf16_t*)p; p += (size_t)D_MODEL * D_MODEL * 2;
    bf16_t* wvh = (bf16_t*)p; p += (size_t)D_MODEL * D_MODEL * 2;
    bf16_t* woh = (bf16_t*)p; p += (size_t)D_MODEL * D_MODEL * 2;
    bf16_t* Qb  = (bf16_t*)p; p += (size_t)64 * T_SEQ * HDIM * 2;
    bf16_t* Kbf = (bf16_t*)p; p += (size_t)64 * T_SEQ * HDIM * 2;
    bf16_t* Vtb = (bf16_t*)p; p += (size_t)64 * T_SEQ * HDIM * 2;
    bf16_t* att = (bf16_t*)p; p += M * D_MODEL * 2;

    const int nx = (int)(M * D_MODEL);
    const int nw = D_MODEL * D_MODEL;
    cvt_f32_bf16<<<dim3((nx + 255) / 256), 256, 0, stream>>>(x, xh, nx);
    cvt_f32_bf16<<<dim3((nw + 255) / 256), 256, 0, stream>>>(Wq, wqh, nw);
    cvt_f32_bf16<<<dim3((nw + 255) / 256), 256, 0, stream>>>(Wk, wkh, nw);
    cvt_f32_bf16<<<dim3((nw + 255) / 256), 256, 0, stream>>>(Wv, wvh, nw);
    cvt_f32_bf16<<<dim3((nw + 255) / 256), 256, 0, stream>>>(Wo, woh, nw);

    const dim3 gg(128, 4), blk(256);
    gemm_bf16_wmma<<<gg, blk, 0, stream>>>(xh, wqh, (void*)Qb, nullptr, 0);
    gemm_bf16_wmma<<<gg, blk, 0, stream>>>(xh, wkh, (void*)Kbf, nullptr, 0);
    gemm_bf16_wmma<<<gg, blk, 0, stream>>>(xh, wvh, (void*)Vtb, nullptr, 1);

    flash_attn_wmma<<<dim3(64, T_SEQ / 128), blk, 0, stream>>>(Qb, Kbf, Vtb, att);

    gemm_bf16_wmma<<<gg, blk, 0, stream>>>(att, woh, (void*)out, bo, 2);
}
